// LSTMModel_5918464934397
// MI455X (gfx1250) — compile-verified, matrix operand order checked by reading, not run on hardware
//
#include <hip/hip_runtime.h>
#include <hip/hip_bf16.h>

// ---------------------------------------------------------------------------
// Fused 2-layer LSTM (B=2048, T=256, I=H=64) + FC head for MI455X (gfx1250).
// wave32 WMMA bf16/f32-accum. Weights + cell state live in VGPRs for the
// whole T loop; h feedback goes through parity-double-buffered LDS staging
// (2 barriers/step). Nonlinearities are branch-free v_exp_f32/v_rcp_f32 with
// the gate bias folded into the exp argument.
// ---------------------------------------------------------------------------

typedef __attribute__((ext_vector_type(16))) __bf16 v16bf;
typedef __attribute__((ext_vector_type(8)))  __bf16 v8bf;
typedef __attribute__((ext_vector_type(8)))  float  v8f;

namespace {
constexpr int kH   = 64;    // hidden (== input) size
constexpr int kT   = 256;   // sequence length
constexpr int kBT  = 16;    // batch rows per block
constexpr int kLD0 = 272;   // A0 row: x[p0]|x[p1]|h0[p0]|h0[p1] + pad (544B)
constexpr int kLD1 = 200;   // A1 row: h0'|h1[p0]|h1[p1] + pad (400B)
constexpr float kL2E = 1.4426950408889634f;
}

// sigmoid(x + b) with nb = -log2e*b:  rcp(1 + exp2(fma(x,-log2e,nb)))
__device__ __forceinline__ float fsig(float x, float nb) {
    const float e = __builtin_amdgcn_exp2f(fmaf(x, -kL2E, nb));
    return __builtin_amdgcn_rcpf(1.0f + e);
}
// tanh(x + b) with nb = -2*log2e*b
__device__ __forceinline__ float ftanh_b(float x, float nb) {
    const float e = __builtin_amdgcn_exp2f(fmaf(x, -2.0f * kL2E, nb));
    return fmaf(__builtin_amdgcn_rcpf(1.0f + e), 2.0f, -1.0f);
}
__device__ __forceinline__ float ftanh(float x) {
    const float e = __builtin_amdgcn_exp2f(x * (-2.0f * kL2E));
    return fmaf(__builtin_amdgcn_rcpf(1.0f + e), 2.0f, -1.0f);
}

// B fragment (16x16 slice of W^T, K=32 deep) from global f32 -> bf16.
// lane L: n = L&15 (W row), khalf = L>>4.
__device__ __forceinline__ v16bf load_bfrag(const float* __restrict__ W,
                                            int row, int kb, int khalf) {
    const float* p = W + row * kH + kb + khalf * 8;
    v16bf r;
#pragma unroll
    for (int i = 0; i < 8; ++i) {
        r[i]     = (__bf16)p[i];
        r[8 + i] = (__bf16)p[16 + i];
    }
    return r;
}

union ufrag { v16bf v; v8bf h[2]; };

__global__ __launch_bounds__(128) void lstm2_fused_kernel(
    const float* __restrict__ x,
    const float* __restrict__ Wih0, const float* __restrict__ Whh0,
    const float* __restrict__ bih0, const float* __restrict__ bhh0,
    const float* __restrict__ Wih1, const float* __restrict__ Whh1,
    const float* __restrict__ bih1, const float* __restrict__ bhh1,
    const float* __restrict__ Wfc,  const float* __restrict__ bfc,
    float* __restrict__ out) {

    __shared__ __attribute__((aligned(16))) __bf16 A0[kBT][kLD0];
    __shared__ __attribute__((aligned(16))) __bf16 A1[kBT][kLD1];
    __shared__ float hlast[kBT][kH + 4];

    const int tid   = threadIdx.x;
    const int lane  = tid & 31;
    const int w     = tid >> 5;          // wave id = hidden-column slice
    const int l16   = lane & 15;
    const int khalf = lane >> 4;
    const int b0    = blockIdx.x * kBT;
    const int ncol  = w * 16 + l16;      // this lane's hidden column (0..63)

    // ---- weight fragments in registers for the whole T loop ---------------
    v16bf Bf0[4][4], Bf1[4][4];          // [gate][kstep]; ks 0,1=Wih, 2,3=Whh
#pragma unroll
    for (int g = 0; g < 4; ++g) {
        const int row = g * kH + ncol;
#pragma unroll
        for (int ks = 0; ks < 2; ++ks) {
            Bf0[g][ks]     = load_bfrag(Wih0, row, ks * 32, khalf);
            Bf0[g][2 + ks] = load_bfrag(Whh0, row, ks * 32, khalf);
            Bf1[g][ks]     = load_bfrag(Wih1, row, ks * 32, khalf);
            Bf1[g][2 + ks] = load_bfrag(Whh1, row, ks * 32, khalf);
        }
    }
    // gate biases pre-scaled into the exp2 argument (g-gate uses tanh scale)
    float bs0[4], bs1[4];
#pragma unroll
    for (int g = 0; g < 4; ++g) {
        const int n = g * kH + ncol;
        const float s = (g == 2) ? (-2.0f * kL2E) : (-kL2E);
        bs0[g] = (bih0[n] + bhh0[n]) * s;
        bs1[g] = (bih1[n] + bhh1[n]) * s;
    }

    // ---- x fetch/stage mapping: rows mrow and mrow+8, 4 floats each --------
    const int mrow = tid >> 4;           // 0..7
    const int k4   = (tid & 15) * 4;
    float4 xva = *(const float4*)&x[((size_t)(b0 + mrow)     * kT + 0) * kH + k4];
    float4 xvb = *(const float4*)&x[((size_t)(b0 + mrow + 8) * kT + 0) * kH + k4];

    // ---- zero-init staging (h0 = h1 = 0), then stage x(0) ------------------
    for (int e = tid; e < kBT * kLD0; e += 128) (&A0[0][0])[e] = (__bf16)0.0f;
    for (int e = tid; e < kBT * kLD1; e += 128) (&A1[0][0])[e] = (__bf16)0.0f;
    __syncthreads();
    {
        union { __bf16 b[4]; unsigned long long u; } pk;
        pk.b[0] = (__bf16)xva.x; pk.b[1] = (__bf16)xva.y;
        pk.b[2] = (__bf16)xva.z; pk.b[3] = (__bf16)xva.w;
        *(unsigned long long*)&A0[mrow][k4] = pk.u;
        pk.b[0] = (__bf16)xvb.x; pk.b[1] = (__bf16)xvb.y;
        pk.b[2] = (__bf16)xvb.z; pk.b[3] = (__bf16)xvb.w;
        *(unsigned long long*)&A0[mrow + 8][k4] = pk.u;
    }
    __syncthreads();

    v8f c0r, c1r;
#pragma unroll
    for (int r = 0; r < 8; ++r) { c0r[r] = 0.0f; c1r[r] = 0.0f; }
    const v8f czero = {0.f, 0.f, 0.f, 0.f, 0.f, 0.f, 0.f, 0.f};
    v8f acc[4];

    // ---- sequential recurrence (2 barriers per step) -----------------------
    for (int t = 0; t < kT; ++t) {
        const int p = t & 1, q = p ^ 1;

        // early global fetch of x(t+1): latency overlapped with both GEMMs
        if (t + 1 < kT) {
            xva = *(const float4*)&x[((size_t)(b0 + mrow)     * kT + (t + 1)) * kH + k4];
            xvb = *(const float4*)&x[((size_t)(b0 + mrow + 8) * kT + (t + 1)) * kH + k4];
        }

        const __bf16* a0x = &A0[l16][p * 64];         // x(t)
        const __bf16* a0h = &A0[l16][128 + p * 64];   // h0(t)
        const __bf16* a1l = &A1[l16][0];              // h0'(t) (after S-mid)
        const __bf16* a1h = &A1[l16][64 + p * 64];    // h1(t)

        // preload layer-1 k-steps 2,3 (h1[p] is stable since t-1)
        ufrag a1k2, a1k3;
        a1k2.h[0] = *(const v8bf*)(a1h + khalf * 8);
        a1k2.h[1] = *(const v8bf*)(a1h + 16 + khalf * 8);
        a1k3.h[0] = *(const v8bf*)(a1h + 32 + khalf * 8);
        a1k3.h[1] = *(const v8bf*)(a1h + 48 + khalf * 8);

        // ---- layer 0 GEMM: gates = [x_t|h0] @ [Wih0|Whh0]^T ---------------
        {
            ufrag a;
            a.h[0] = *(const v8bf*)(a0x + khalf * 8);
            a.h[1] = *(const v8bf*)(a0x + 16 + khalf * 8);
#pragma unroll
            for (int g = 0; g < 4; ++g)
                acc[g] = __builtin_amdgcn_wmma_f32_16x16x32_bf16(
                    false, a.v, false, Bf0[g][0], (short)0, czero, false, false);
            a.h[0] = *(const v8bf*)(a0x + 32 + khalf * 8);
            a.h[1] = *(const v8bf*)(a0x + 48 + khalf * 8);
#pragma unroll
            for (int g = 0; g < 4; ++g)
                acc[g] = __builtin_amdgcn_wmma_f32_16x16x32_bf16(
                    false, a.v, false, Bf0[g][1], (short)0, acc[g], false, false);
            a.h[0] = *(const v8bf*)(a0h + khalf * 8);
            a.h[1] = *(const v8bf*)(a0h + 16 + khalf * 8);
#pragma unroll
            for (int g = 0; g < 4; ++g)
                acc[g] = __builtin_amdgcn_wmma_f32_16x16x32_bf16(
                    false, a.v, false, Bf0[g][2], (short)0, acc[g], false, false);
            a.h[0] = *(const v8bf*)(a0h + 32 + khalf * 8);
            a.h[1] = *(const v8bf*)(a0h + 48 + khalf * 8);
#pragma unroll
            for (int g = 0; g < 4; ++g)
                acc[g] = __builtin_amdgcn_wmma_f32_16x16x32_bf16(
                    false, a.v, false, Bf0[g][3], (short)0, acc[g], false, false);
        }
        // ---- layer 0 cell update (registers), branch-free -----------------
#pragma unroll
        for (int r = 0; r < 8; ++r) {
            const float ig = fsig(acc[0][r], bs0[0]);
            const float fg = fsig(acc[1][r], bs0[1]);
            const float gg = ftanh_b(acc[2][r], bs0[2]);
            const float og = fsig(acc[3][r], bs0[3]);
            const float c  = fmaf(fg, c0r[r], ig * gg);
            c0r[r] = c;
            const __bf16 hb = (__bf16)(og * ftanh(c));
            const int m = r + 8 * khalf;
            A0[m][128 + q * 64 + ncol] = hb;   // h0 for step t+1
            A1[m][ncol]                = hb;   // h0' for layer 1 (this step)
        }
        __syncthreads();                       // S-mid: h0' visible

        // ---- layer 1 GEMM: ks2,ks3 use preloaded frags, then ks0,ks1 ------
        {
#pragma unroll
            for (int g = 0; g < 4; ++g)
                acc[g] = __builtin_amdgcn_wmma_f32_16x16x32_bf16(
                    false, a1k2.v, false, Bf1[g][2], (short)0, czero, false, false);
#pragma unroll
            for (int g = 0; g < 4; ++g)
                acc[g] = __builtin_amdgcn_wmma_f32_16x16x32_bf16(
                    false, a1k3.v, false, Bf1[g][3], (short)0, acc[g], false, false);
            ufrag a;
            a.h[0] = *(const v8bf*)(a1l + khalf * 8);
            a.h[1] = *(const v8bf*)(a1l + 16 + khalf * 8);
#pragma unroll
            for (int g = 0; g < 4; ++g)
                acc[g] = __builtin_amdgcn_wmma_f32_16x16x32_bf16(
                    false, a.v, false, Bf1[g][0], (short)0, acc[g], false, false);
            a.h[0] = *(const v8bf*)(a1l + 32 + khalf * 8);
            a.h[1] = *(const v8bf*)(a1l + 48 + khalf * 8);
#pragma unroll
            for (int g = 0; g < 4; ++g)
                acc[g] = __builtin_amdgcn_wmma_f32_16x16x32_bf16(
                    false, a.v, false, Bf1[g][1], (short)0, acc[g], false, false);
        }
        // ---- layer 1 cell update ------------------------------------------
#pragma unroll
        for (int r = 0; r < 8; ++r) {
            const float ig = fsig(acc[0][r], bs1[0]);
            const float fg = fsig(acc[1][r], bs1[1]);
            const float gg = ftanh_b(acc[2][r], bs1[2]);
            const float og = fsig(acc[3][r], bs1[3]);
            const float c  = fmaf(fg, c1r[r], ig * gg);
            c1r[r] = c;
            const float h  = og * ftanh(c);
            const int m = r + 8 * khalf;
            A1[m][64 + q * 64 + ncol] = (__bf16)h;   // h1 for step t+1
            if (t == kT - 1) hlast[m][ncol] = h;     // f32 for FC head
        }
        // stage x(t+1) into parity-q slot (regs were fetched at loop top)
        if (t + 1 < kT) {
            union { __bf16 b[4]; unsigned long long u; } pk;
            pk.b[0] = (__bf16)xva.x; pk.b[1] = (__bf16)xva.y;
            pk.b[2] = (__bf16)xva.z; pk.b[3] = (__bf16)xva.w;
            *(unsigned long long*)&A0[mrow][q * 64 + k4] = pk.u;
            pk.b[0] = (__bf16)xvb.x; pk.b[1] = (__bf16)xvb.y;
            pk.b[2] = (__bf16)xvb.z; pk.b[3] = (__bf16)xvb.w;
            *(unsigned long long*)&A0[mrow + 8][q * 64 + k4] = pk.u;
        }
        __syncthreads();                       // S-end: publish h0,h1,x(t+1)
    }

    // ---- FC head: out = h1_last @ Wfc^T + bfc (16 rows x 3 outputs) --------
    if (tid < kBT * 3) {
        const int m = tid / 3, o = tid % 3;
        float s = bfc[o];
#pragma unroll
        for (int k = 0; k < kH; ++k) s += hlast[m][k] * Wfc[o * kH + k];
        out[(size_t)(b0 + m) * 3 + o] = s;
    }
}

extern "C" void kernel_launch(void* const* d_in, const int* in_sizes, int n_in,
                              void* d_out, int out_size, void* d_ws, size_t ws_size,
                              hipStream_t stream) {
    (void)n_in; (void)d_ws; (void)ws_size; (void)out_size;
    const float* x    = (const float*)d_in[0];
    const float* Wih0 = (const float*)d_in[1];
    const float* Whh0 = (const float*)d_in[2];
    const float* bih0 = (const float*)d_in[3];
    const float* bhh0 = (const float*)d_in[4];
    const float* Wih1 = (const float*)d_in[5];
    const float* Whh1 = (const float*)d_in[6];
    const float* bih1 = (const float*)d_in[7];
    const float* bhh1 = (const float*)d_in[8];
    const float* Wfc  = (const float*)d_in[9];
    const float* bfc  = (const float*)d_in[10];
    float* out = (float*)d_out;

    const int Bsz  = in_sizes[0] / (kT * kH);  // 2048
    const int grid = Bsz / kBT;                // 16 batch rows per block

    lstm2_fused_kernel<<<grid, 128, 0, stream>>>(
        x, Wih0, Whh0, bih0, bhh0, Wih1, Whh1, bih1, bhh1, Wfc, bfc, out);
}
